// SharedOnlyMLP_59433757442706
// MI455X (gfx1250) — matrix operand order repaired and checked by reading, not compiled, not run on hardware
//
#include <hip/hip_runtime.h>

typedef __attribute__((ext_vector_type(16))) _Float16 v16h;
typedef __attribute__((ext_vector_type(8)))  _Float16 v8h;
typedef __attribute__((ext_vector_type(8)))  float    v8f;
typedef __attribute__((ext_vector_type(4)))  unsigned int v4u;
typedef __attribute__((ext_vector_type(8)))  int      v8i;
typedef __attribute__((ext_vector_type(4)))  int      v4i;

#define D_MODEL 4096
#define D_FF    11008
#define NTOK    8192   // 4 * 2048

#define BM 128         // block tile M (tokens)
#define BN 64          // block tile N (output features)
#define TK 64          // K elements staged per LDS buffer (2 WMMA depths)

// ---- workspace layout (bytes) ----
#define XH_OFF  ((size_t)0)                      // 8192*4096 f16  = 64 MiB
#define GW_OFF  ((size_t)67108864)               // 11008*4096 f16 = 86 MiB
#define UW_OFF  ((size_t)157286400)
#define DW_OFF  ((size_t)247463936)
#define H_OFF   ((size_t)337641472)              // 8192*11008 f16 = 172 MiB

#if __has_builtin(__builtin_amdgcn_tensor_load_to_lds)
#define USE_TDM 1
#else
#define USE_TDM 0
#endif

// ---------------- prep: int32 (int8 values) -> f16 (exact) ----------------
__global__ void cvt_i32_to_f16(const int* __restrict__ w, _Float16* __restrict__ o, long n) {
    long i = ((long)blockIdx.x * blockDim.x + threadIdx.x) * 8;
    if (i + 8 > n) return;
    int4 a = *(const int4*)(w + i);
    int4 b = *(const int4*)(w + i + 4);
    v8h r;
    r[0] = (_Float16)(float)a.x; r[1] = (_Float16)(float)a.y;
    r[2] = (_Float16)(float)a.z; r[3] = (_Float16)(float)a.w;
    r[4] = (_Float16)(float)b.x; r[5] = (_Float16)(float)b.y;
    r[6] = (_Float16)(float)b.z; r[7] = (_Float16)(float)b.w;
    *(v8h*)(o + i) = r;
}

// ---------------- prep: f32 -> f16 ----------------
__global__ void cvt_f32_to_f16(const float* __restrict__ x, _Float16* __restrict__ o, long n) {
    long i = ((long)blockIdx.x * blockDim.x + threadIdx.x) * 8;
    if (i + 8 > n) return;
    float4 a = *(const float4*)(x + i);
    float4 b = *(const float4*)(x + i + 4);
    v8h r;
    r[0] = (_Float16)a.x; r[1] = (_Float16)a.y; r[2] = (_Float16)a.z; r[3] = (_Float16)a.w;
    r[4] = (_Float16)b.x; r[5] = (_Float16)b.y; r[6] = (_Float16)b.z; r[7] = (_Float16)b.w;
    *(v8h*)(o + i) = r;
}

// ---- WMMA fragment assembly per CDNA5 ISA lane layouts ----
// LDS tiles are row-major [rows][TK] halves; kh selects the 32-wide K-half.
// A (16x32 f16): lanes 0-15 row M=lane, K 0..7 & 16..23; lanes 16-31 K 8..15 & 24..31.
__device__ __forceinline__ v16h load_a_frag(const _Float16* As, int m_local, int kh, int lane) {
    int ksel = (lane & 16) ? 8 : 0;
    const _Float16* p = As + m_local * TK + kh * 32 + ksel;
    union { v16h v; v8h h[2]; } u;
    u.h[0] = *(const v8h*)(p);
    u.h[1] = *(const v8h*)(p + 16);
    return u.v;
}
// B (32x16 f16): lanes 0-15 col N=lane, K 0..7 & 16..23; lanes 16-31 col N=lane-16, K 8..15 & 24..31.
__device__ __forceinline__ v16h load_b_frag(const _Float16* Ws, int f_local, int kh, int lane) {
    int kb = (lane & 16) ? 8 : 0;
    const _Float16* p = Ws + f_local * TK + kh * 32 + kb;
    union { v16h v; v8h h[2]; } u;
    u.h[0] = *(const v8h*)(p);
    u.h[1] = *(const v8h*)(p + 16);
    return u.v;
}

__device__ __forceinline__ v8f wmma_f16(v16h a, v16h b, v8f c) {
    return __builtin_amdgcn_wmma_f32_16x16x32_f16(false, a, false, b, (short)0, c, false, false);
}

// ---- Tensor Data Mover: 2-D tile (tile_w x tile_h, f16) global -> LDS ----
// D# layout per CDNA5 ISA.
__device__ __forceinline__ unsigned lds_addr32(const void* p) {
    return (unsigned)(unsigned long long)(uintptr_t)p;   // low 32 bits = LDS offset
}

#if USE_TDM
__device__ __forceinline__ void tdm_load_2d(const void* lds_dst, const void* gsrc,
                                            unsigned row_len, unsigned nrows,
                                            unsigned row_stride,
                                            unsigned tile_w, unsigned tile_h) {
    unsigned long long ga = (unsigned long long)(uintptr_t)gsrc;
    v4u g0;
    g0[0] = 1u;                                            // count=1 (valid), user mode
    g0[1] = lds_addr32(lds_dst);                           // lds_addr
    g0[2] = (unsigned)(ga & 0xFFFFFFFFu);                  // global_addr[31:0]
    g0[3] = (unsigned)((ga >> 32) & 0x1FFFFFFu) | (2u << 30); // global_addr[56:32] | type=2
    v8i g1;
    g1[0] = (int)(1u << 16);                               // data_size=1 -> 2 bytes
    g1[1] = (int)((row_len & 0xFFFFu) << 16);              // tensor_dim0[15:0]
    g1[2] = (int)(((row_len >> 16) & 0xFFFFu) | ((nrows & 0xFFFFu) << 16));
    g1[3] = (int)(((nrows >> 16) & 0xFFFFu) | ((tile_w & 0xFFFFu) << 16)); // tile_dim0
    g1[4] = (int)(tile_h & 0xFFFFu);                       // tile_dim1 (tile_dim2=0)
    g1[5] = (int)row_stride;                               // tensor_dim0_stride[31:0]
    g1[6] = 0;
    g1[7] = 0;
    v4i z = {0, 0, 0, 0};
#if defined(__clang_major__) && (__clang_major__ >= 23)
    v8i z8 = {0, 0, 0, 0, 0, 0, 0, 0};
    __builtin_amdgcn_tensor_load_to_lds(g0, g1, z, z, z8, 0);
#else
    __builtin_amdgcn_tensor_load_to_lds(g0, g1, z, z, 0);
#endif
}
#define TDM_WAIT() __builtin_amdgcn_s_wait_tensorcnt(0)
#else
// fallback: cooperative synchronous tile load (rows x TK halves)
__device__ __forceinline__ void coop_load_tile(_Float16* dst, const _Float16* src,
                                               int rows, size_t src_stride, int tid) {
    int per = rows * (TK / 8);
    for (int i = tid; i < per; i += 256) {
        int r = i / (TK / 8); int c8 = (i % (TK / 8)) * 8;
        *(v8h*)(dst + r * TK + c8) = *(const v8h*)(src + (size_t)r * src_stride + c8);
    }
}
#endif

// ---------------- fused gate/up GEMM + SiLU*up -> h (f16) ----------------
// grid.x = D_FF/BN, grid.y = NTOK/BM; 256 threads = 8 waves (4 M-rows x 2 N-cols),
// each wave owns a 32x32 output tile of BOTH gate and up (8 f32 accumulators).
// One staged buffer = K depth 64 -> 16 WMMAs per barrier pair per wave.
__global__ __launch_bounds__(256)
void gemm_gateup_silu(const _Float16* __restrict__ xh,
                      const _Float16* __restrict__ gwh,
                      const _Float16* __restrict__ uwh,
                      const float* __restrict__ gate_s,
                      const float* __restrict__ up_s,
                      _Float16* __restrict__ hbuf) {
    __shared__ __align__(16) _Float16 As[2][BM * TK];   // 2 x 16 KiB
    __shared__ __align__(16) _Float16 Wg[2][BN * TK];   // 2 x 8 KiB
    __shared__ __align__(16) _Float16 Wu[2][BN * TK];   // 2 x 8 KiB
    __shared__ __align__(16) _Float16 Ht[BM * BN];      // 16 KiB epilogue staging

    const int tid  = threadIdx.x;
    const int lane = tid & 31;
    const int wave = tid >> 5;
    const int wm   = wave & 3;   // wave M-row (0..3) -> 32 rows each
    const int wn   = wave >> 2;  // wave N-col (0..1) -> 32 cols each

    const int fblock = blockIdx.x * BN;
    const int mbase  = blockIdx.y * BM;

    const int mloc0 = wm * 32 + (lane & 15);
    const int mloc1 = mloc0 + 16;
    const int floc0 = wn * 32 + (lane & 15);
    const int floc1 = floc0 + 16;

    v8f g00 = {}, g01 = {}, g10 = {}, g11 = {};
    v8f u00 = {}, u01 = {}, u10 = {}, u11 = {};

#if USE_TDM
    if (wave == 0) {
        tdm_load_2d(&As[0][0], xh  + (size_t)mbase  * D_MODEL, D_MODEL, NTOK, D_MODEL, TK, BM);
        tdm_load_2d(&Wg[0][0], gwh + (size_t)fblock * D_MODEL, D_MODEL, D_FF, D_MODEL, TK, BN);
        tdm_load_2d(&Wu[0][0], uwh + (size_t)fblock * D_MODEL, D_MODEL, D_FF, D_MODEL, TK, BN);
    }
#endif

    for (int k0 = 0; k0 < D_MODEL; k0 += TK) {
        const int pb = (k0 / TK) & 1;
#if USE_TDM
        TDM_WAIT();                                // no-op for non-issuing waves
        __syncthreads();
        if (wave == 0 && k0 + TK < D_MODEL) {      // DMA next tiles into other buffer
            const int nb = pb ^ 1;
            tdm_load_2d(&As[nb][0], xh  + (size_t)mbase  * D_MODEL + k0 + TK, D_MODEL, NTOK, D_MODEL, TK, BM);
            tdm_load_2d(&Wg[nb][0], gwh + (size_t)fblock * D_MODEL + k0 + TK, D_MODEL, D_FF, D_MODEL, TK, BN);
            tdm_load_2d(&Wu[nb][0], uwh + (size_t)fblock * D_MODEL + k0 + TK, D_MODEL, D_FF, D_MODEL, TK, BN);
        }
#else
        __syncthreads();
        coop_load_tile(&As[pb][0], xh  + (size_t)mbase  * D_MODEL + k0, BM, D_MODEL, tid);
        coop_load_tile(&Wg[pb][0], gwh + (size_t)fblock * D_MODEL + k0, BN, D_MODEL, tid);
        coop_load_tile(&Wu[pb][0], uwh + (size_t)fblock * D_MODEL + k0, BN, D_MODEL, tid);
        __syncthreads();
#endif
        const _Float16* Ap = &As[pb][0];
        const _Float16* Gp = &Wg[pb][0];
        const _Float16* Up = &Wu[pb][0];

        #pragma unroll
        for (int kh = 0; kh < 2; ++kh) {
            v16h a0 = load_a_frag(Ap, mloc0, kh, lane);
            v16h a1 = load_a_frag(Ap, mloc1, kh, lane);
            v16h b;
            b = load_b_frag(Gp, floc0, kh, lane);
            g00 = wmma_f16(a0, b, g00);  g10 = wmma_f16(a1, b, g10);
            b = load_b_frag(Gp, floc1, kh, lane);
            g01 = wmma_f16(a0, b, g01);  g11 = wmma_f16(a1, b, g11);
            b = load_b_frag(Up, floc0, kh, lane);
            u00 = wmma_f16(a0, b, u00);  u10 = wmma_f16(a1, b, u10);
            b = load_b_frag(Up, floc1, kh, lane);
            u01 = wmma_f16(a0, b, u01);  u11 = wmma_f16(a1, b, u11);
        }
        __syncthreads();
    }

    // epilogue: scale, SiLU(gate)*up, transpose via LDS, coalesced f16 rows out
    {
        const int lhalf = lane >> 4;
        const int lcol  = lane & 15;
        #pragma unroll
        for (int s = 0; s < 2; ++s) {
            #pragma unroll
            for (int t = 0; t < 2; ++t) {
                const int fc = wn * 32 + t * 16 + lcol;
                const float sg = gate_s[fblock + fc];
                const float su = up_s[fblock + fc];
                const v8f g = s ? (t ? g11 : g10) : (t ? g01 : g00);
                const v8f u = s ? (t ? u11 : u10) : (t ? u01 : u00);
                #pragma unroll
                for (int r = 0; r < 8; ++r) {
                    const int row = wm * 32 + s * 16 + r + 8 * lhalf;
                    const float gv = g[r] * sg;
                    const float uv = u[r] * su;
                    const float hv = (gv / (1.0f + __expf(-gv))) * uv;
                    Ht[row * BN + fc] = (_Float16)hv;
                }
            }
        }
    }
    __syncthreads();
    #pragma unroll
    for (int i = 0; i < 4; ++i) {
        const int idx = tid + i * 256;           // BM*BN/8 = 1024 v8h segments
        const int row = idx >> 3;
        const int c8  = (idx & 7) * 8;
        *(v8h*)(hbuf + (size_t)(mbase + row) * D_FF + fblock + c8) = *(const v8h*)(Ht + row * BN + c8);
    }
}

// ---------------- down projection: out = h @ (down_w*down_s)^T (f32 out) ----------------
// grid.x = D_MODEL/BN, grid.y = NTOK/BM; each wave: 32x32 output (4 accumulators)
__global__ __launch_bounds__(256)
void gemm_down(const _Float16* __restrict__ hbuf,
               const _Float16* __restrict__ dwh,
               const float* __restrict__ down_s,
               float* __restrict__ out) {
    __shared__ __align__(16) _Float16 As[2][BM * TK];   // 2 x 16 KiB
    __shared__ __align__(16) _Float16 Wd[2][BN * TK];   // 2 x 8 KiB

    const int tid  = threadIdx.x;
    const int lane = tid & 31;
    const int wave = tid >> 5;
    const int wm   = wave & 3;
    const int wn   = wave >> 2;

    const int dblock = blockIdx.x * BN;
    const int mbase  = blockIdx.y * BM;

    const int mloc0 = wm * 32 + (lane & 15);
    const int mloc1 = mloc0 + 16;
    const int floc0 = wn * 32 + (lane & 15);
    const int floc1 = floc0 + 16;

    v8f a00 = {}, a01 = {}, a10 = {}, a11 = {};

#if USE_TDM
    if (wave == 0) {
        tdm_load_2d(&As[0][0], hbuf + (size_t)mbase  * D_FF, D_FF, NTOK,    D_FF, TK, BM);
        tdm_load_2d(&Wd[0][0], dwh  + (size_t)dblock * D_FF, D_FF, D_MODEL, D_FF, TK, BN);
    }
#endif

    for (int k0 = 0; k0 < D_FF; k0 += TK) {
        const int pb = (k0 / TK) & 1;
#if USE_TDM
        TDM_WAIT();
        __syncthreads();
        if (wave == 0 && k0 + TK < D_FF) {
            const int nb = pb ^ 1;
            tdm_load_2d(&As[nb][0], hbuf + (size_t)mbase  * D_FF + k0 + TK, D_FF, NTOK,    D_FF, TK, BM);
            tdm_load_2d(&Wd[nb][0], dwh  + (size_t)dblock * D_FF + k0 + TK, D_FF, D_MODEL, D_FF, TK, BN);
        }
#else
        __syncthreads();
        coop_load_tile(&As[pb][0], hbuf + (size_t)mbase  * D_FF + k0, BM, D_FF, tid);
        coop_load_tile(&Wd[pb][0], dwh  + (size_t)dblock * D_FF + k0, BN, D_FF, tid);
        __syncthreads();
#endif
        const _Float16* Ap = &As[pb][0];
        const _Float16* Wp = &Wd[pb][0];

        #pragma unroll
        for (int kh = 0; kh < 2; ++kh) {
            v16h a0 = load_a_frag(Ap, mloc0, kh, lane);
            v16h a1 = load_a_frag(Ap, mloc1, kh, lane);
            v16h b;
            b = load_b_frag(Wp, floc0, kh, lane);
            a00 = wmma_f16(a0, b, a00);  a10 = wmma_f16(a1, b, a10);
            b = load_b_frag(Wp, floc1, kh, lane);
            a01 = wmma_f16(a0, b, a01);  a11 = wmma_f16(a1, b, a11);
        }
        __syncthreads();
    }

    const int lhalf = lane >> 4;
    const int lcol  = lane & 15;
    #pragma unroll
    for (int s = 0; s < 2; ++s) {
        #pragma unroll
        for (int t = 0; t < 2; ++t) {
            const int d = dblock + wn * 32 + t * 16 + lcol;
            const float sd = down_s[d];
            const v8f a = s ? (t ? a11 : a10) : (t ? a01 : a00);
            #pragma unroll
            for (int r = 0; r < 8; ++r) {
                const int row = mbase + wm * 32 + s * 16 + r + 8 * lhalf;
                out[(size_t)row * D_MODEL + d] = a[r] * sd;
            }
        }
    }
}

extern "C" void kernel_launch(void* const* d_in, const int* in_sizes, int n_in,
                              void* d_out, int out_size, void* d_ws, size_t ws_size,
                              hipStream_t stream) {
    const float* x       = (const float*)d_in[0];
    const int*   gate_w  = (const int*)d_in[1];
    const float* gate_s  = (const float*)d_in[2];
    const int*   up_w    = (const int*)d_in[3];
    const float* up_s    = (const float*)d_in[4];
    const int*   down_w  = (const int*)d_in[5];
    const float* down_s  = (const float*)d_in[6];
    float*       out     = (float*)d_out;

    char* ws = (char*)d_ws;
    _Float16* xh   = (_Float16*)(ws + XH_OFF);
    _Float16* gwh  = (_Float16*)(ws + GW_OFF);
    _Float16* uwh  = (_Float16*)(ws + UW_OFF);
    _Float16* dwh  = (_Float16*)(ws + DW_OFF);
    _Float16* hbuf = (_Float16*)(ws + H_OFF);

    const long n_x = (long)NTOK * D_MODEL;
    const long n_w = (long)D_FF * D_MODEL;

    cvt_f32_to_f16<<<(int)(n_x / 8 / 256), 256, 0, stream>>>(x, xh, n_x);
    cvt_i32_to_f16<<<(int)(n_w / 8 / 256), 256, 0, stream>>>(gate_w, gwh, n_w);
    cvt_i32_to_f16<<<(int)(n_w / 8 / 256), 256, 0, stream>>>(up_w,   uwh, n_w);
    cvt_i32_to_f16<<<(int)(n_w / 8 / 256), 256, 0, stream>>>(down_w, dwh, n_w);

    dim3 gridGU(D_FF / BN, NTOK / BM);      // 172 x 64
    gemm_gateup_silu<<<gridGU, 256, 0, stream>>>(xh, gwh, uwh, gate_s, up_s, hbuf);

    dim3 gridD(D_MODEL / BN, NTOK / BM);    // 64 x 64
    gemm_down<<<gridD, 256, 0, stream>>>(hbuf, dwh, down_s, out);
}
